// LSTM_5506148074102
// MI455X (gfx1250) — compile-verified, matrix operand order checked by reading, not run on hardware
//
#include <hip/hip_runtime.h>

// ---------------- problem constants ----------------
#define BATCH 4096
#define TSTEPS 32
#define EMB 300
#define HID 256
#define G4 1024          // 4*HID
#define NCLS 10
#define KE 320           // EMB padded to multiple of 32
#define KTOT 576         // KE + HID (18 chunks of K=32)
#define NKC 18           // KTOT / 32
#define TB 32            // batch tile per workgroup
#define ASTR 584         // LDS act row stride (halves), padded
#define GSTR 1032        // LDS gates row stride (floats), padded

typedef __attribute__((ext_vector_type(16))) _Float16 v16h;
typedef __attribute__((ext_vector_type(8)))  _Float16 v8h;
typedef __attribute__((ext_vector_type(4)))  _Float16 v4h;
typedef __attribute__((ext_vector_type(8)))  float    v8f;

// Load one 16-lane-row fragment for V_WMMA_F32_16X16X32_F16.
// Per ISA layout: lane holds row (lane&15); half = lane>>4 selects
// K = 8h..8h+7 (VGPR0-3) and K = 16+8h..16+8h+7 (VGPR4-7) of the 32-K chunk.
// base points at (row start + kc*32) in halves.
static __device__ inline v16h load_frag(const _Float16* base, int half) {
  v8h lo = *reinterpret_cast<const v8h*>(base + 8 * half);
  v8h hi = *reinterpret_cast<const v8h*>(base + 16 + 8 * half);
  v16h r;
#pragma unroll
  for (int i = 0; i < 8; ++i) { r[i] = lo[i]; r[8 + i] = hi[i]; }
  return r;
}

static __device__ inline float sigmoidf_(float x) {
  return 1.0f / (1.0f + __expf(-x));
}

// ---------------- phase 0: pack weights to f16 fused layout ----------------
// Wfused[dir]: row-major (1024, 576) f16; cols [0,300)=Wih, [300,320)=0,
// [320,576)=Whh.  bias[dir][g] = bih[g]+bhh[g] (f32).
__global__ void lstm_prep_kernel(const float* __restrict__ WihF, const float* __restrict__ WhhF,
                                 const float* __restrict__ bihF, const float* __restrict__ bhhF,
                                 const float* __restrict__ WihB, const float* __restrict__ WhhB,
                                 const float* __restrict__ bihB, const float* __restrict__ bhhB,
                                 _Float16* __restrict__ Wf, float* __restrict__ bias) {
  long idx = (long)blockIdx.x * blockDim.x + threadIdx.x;
  const long per = (long)G4 * KTOT;
  if (idx < 2 * per) {
    int d = (int)(idx / per);
    long r = idx % per;
    int g  = (int)(r / KTOT);
    int cc = (int)(r % KTOT);
    const float* Wih = d ? WihB : WihF;
    const float* Whh = d ? WhhB : WhhF;
    float v;
    if (cc < EMB)      v = Wih[g * EMB + cc];
    else if (cc < KE)  v = 0.0f;
    else               v = Whh[g * HID + (cc - KE)];
    Wf[idx] = (_Float16)v;
  }
  if (idx < 2 * G4) {
    int d = (int)(idx >> 10), g = (int)(idx & (G4 - 1));
    bias[idx] = d ? (bihB[g] + bhhB[g]) : (bihF[g] + bhhF[g]);
  }
}

// ---------------- phase 1: persistent fused BiLSTM kernel ----------------
// grid = (BATCH/TB, 2 dirs), block = 256 threads (8 waves).
// Each workgroup runs all T steps for a 32-row batch tile of one direction,
// computing gates = [e_t | h] @ Wfused^T + b with WMMA, then the pointwise
// LSTM cell, keeping c and running max(relu(h)) in registers.
// Loop order: A-fragments for both 16-row M-tiles preloaded once per step;
// each N-tile streams its weight fragments from L2 exactly once, feeding
// two independent accumulator chains (m0, m1) interleaved per K-step.
__global__ __launch_bounds__(256, 1)
void lstm_dir_kernel(const int* __restrict__ x, const float* __restrict__ emb,
                     const _Float16* __restrict__ Wfused, const float* __restrict__ bias,
                     float* __restrict__ pooled) {
  __shared__ _Float16 s_act[TB * ASTR];   // [e_t | h] tile, f16   (~37 KB)
  __shared__ float    s_gates[TB * GSTR]; // gate tile, f32        (~132 KB)
  __shared__ int      s_tok[TB];

  const int dir  = blockIdx.y;
  const int b0   = blockIdx.x * TB;
  const int tid  = threadIdx.x;
  const int lane = tid & 31;
  const int wave = tid >> 5;
  const int col  = lane & 15;
  const int half = lane >> 4;

  const _Float16* W  = Wfused + (long)dir * G4 * KTOT;
  const float*    bs = bias + dir * G4;

  // zero h region of activation tile (h0 = 0)
  for (int idx = tid; idx < TB * HID; idx += 256) {
    int r = idx / HID, c = idx % HID;
    s_act[r * ASTR + KE + c] = (_Float16)0.0f;
  }

  // each thread owns 32 (b, j) cells: b = tid>>3, j = (tid&7)*32 + u
  const int ob  = tid >> 3;
  const int oj0 = (tid & 7) * 32;
  float c_reg[32], pm[32];
#pragma unroll
  for (int u = 0; u < 32; ++u) { c_reg[u] = 0.0f; pm[u] = 0.0f; }

  for (int ts = 0; ts < TSTEPS; ++ts) {
    const int t = dir ? (TSTEPS - 1 - ts) : ts;
    if (tid < TB) s_tok[tid] = x[(b0 + tid) * TSTEPS + t];
    __syncthreads();

    // gather embeddings -> f16 act tile, vectorized: 300 f32 = 75 float4
    // per row; slots 75..79 are the zero pad (cols 300..319).
    for (int idx = tid; idx < TB * 80; idx += 256) {
      int r = idx / 80, s = idx % 80;
      v4h o;
      if (s < 75) {
        const float4 v =
            *reinterpret_cast<const float4*>(emb + (long)s_tok[r] * EMB + 4 * s);
        o[0] = (_Float16)v.x; o[1] = (_Float16)v.y;
        o[2] = (_Float16)v.z; o[3] = (_Float16)v.w;
      } else {
        o[0] = (_Float16)0.f; o[1] = (_Float16)0.f;
        o[2] = (_Float16)0.f; o[3] = (_Float16)0.f;
      }
      *reinterpret_cast<v4h*>(s_act + r * ASTR + 4 * s) = o;
    }
    __syncthreads();

    // preload A fragments for both M-tiles (rows 0-15 and 16-31)
    v16h a0[NKC], a1[NKC];
    {
      const _Float16* arow0 = s_act + col * ASTR;
      const _Float16* arow1 = s_act + (16 + col) * ASTR;
#pragma unroll
      for (int kc = 0; kc < NKC; ++kc) {
        a0[kc] = load_frag(arow0 + kc * 32, half);
        a1[kc] = load_frag(arow1 + kc * 32, half);
      }
    }

    // GEMM: gates(32 x 1024) = act(32 x 576) @ W^T + b, via 16x16x32 WMMA.
    // One pass over the weight stream per N-tile feeds two accumulators.
    for (int ns = 0; ns < 8; ++ns) {
      const int nt = wave * 8 + ns;
      const float bv = bs[nt * 16 + col];
      v8f acc0, acc1;
#pragma unroll
      for (int r = 0; r < 8; ++r) { acc0[r] = bv; acc1[r] = bv; }
      const _Float16* brow = W + (long)(nt * 16 + col) * KTOT;
#pragma unroll
      for (int kc = 0; kc < NKC; ++kc) {
        v16h b = load_frag(brow + kc * 32, half);
        acc0 = __builtin_amdgcn_wmma_f32_16x16x32_f16(
            false, a0[kc], false, b, (short)0, acc0, false, false);
        acc1 = __builtin_amdgcn_wmma_f32_16x16x32_f16(
            false, a1[kc], false, b, (short)0, acc1, false, false);
      }
      float* gout = s_gates + nt * 16 + col;
#pragma unroll
      for (int r = 0; r < 8; ++r) {
        gout[(half * 8 + r) * GSTR]      = acc0[r];
        gout[(16 + half * 8 + r) * GSTR] = acc1[r];
      }
    }
    __syncthreads();

    // pointwise LSTM cell + fold relu+max-over-time; write h back (f16)
    const float* gr = s_gates + ob * GSTR;
#pragma unroll
    for (int u = 0; u < 32; ++u) {
      int j   = oj0 + u;
      float ig = sigmoidf_(gr[j]);
      float fg = sigmoidf_(gr[HID + j]);
      float gg = tanhf(gr[2 * HID + j]);
      float og = sigmoidf_(gr[3 * HID + j]);
      float cv = fg * c_reg[u] + ig * gg;
      c_reg[u] = cv;
      float h = og * tanhf(cv);
      pm[u] = fmaxf(pm[u], h);          // == max over t of relu(h)
      s_act[ob * ASTR + KE + j] = (_Float16)h;
    }
    __syncthreads();
  }

  // pooled: (B, 2H), forward dir in cols [0,256), backward in [256,512)
  float* pout = pooled + (long)(b0 + ob) * (2 * HID) + dir * HID + oj0;
#pragma unroll
  for (int u = 0; u < 32; ++u) pout[u] = pm[u];
}

// ---------------- phase 2: tiny FC (B x 512) @ (512 x 10) ----------------
__global__ void lstm_fc_kernel(const float* __restrict__ pooled,
                               const float* __restrict__ fcW,
                               const float* __restrict__ fcb,
                               float* __restrict__ out) {
  int idx = blockIdx.x * blockDim.x + threadIdx.x;
  if (idx >= BATCH * NCLS) return;
  int b = idx / NCLS, n = idx % NCLS;
  const float* p = pooled + (long)b * (2 * HID);
  const float* w = fcW + n * (2 * HID);
  float acc = fcb[n];
#pragma unroll 8
  for (int k = 0; k < 2 * HID; ++k) acc += p[k] * w[k];
  out[idx] = acc;
}

// ---------------- launcher ----------------
extern "C" void kernel_launch(void* const* d_in, const int* in_sizes, int n_in,
                              void* d_out, int out_size, void* d_ws, size_t ws_size,
                              hipStream_t stream) {
  const int*   x     = (const int*)d_in[0];
  const float* emb   = (const float*)d_in[1];
  const float* WihF  = (const float*)d_in[2];
  const float* WhhF  = (const float*)d_in[3];
  const float* bihF  = (const float*)d_in[4];
  const float* bhhF  = (const float*)d_in[5];
  const float* WihB  = (const float*)d_in[6];
  const float* WhhB  = (const float*)d_in[7];
  const float* bihB  = (const float*)d_in[8];
  const float* bhhB  = (const float*)d_in[9];
  const float* fcW   = (const float*)d_in[10];
  const float* fcb   = (const float*)d_in[11];
  float* out = (float*)d_out;

  // workspace layout (~10.8 MB)
  char* ws = (char*)d_ws;
  const size_t wf_bytes   = (size_t)2 * G4 * KTOT * sizeof(_Float16); // 2,359,296
  const size_t bias_bytes = (size_t)2 * G4 * sizeof(float);           // 8,192
  _Float16* Wf     = (_Float16*)ws;
  float*    bias   = (float*)(ws + wf_bytes);
  float*    pooled = (float*)(ws + wf_bytes + bias_bytes);            // B*512 f32

  // phase 0: weight packing
  {
    long total = 2L * G4 * KTOT;
    int blocks = (int)((total + 255) / 256);
    lstm_prep_kernel<<<blocks, 256, 0, stream>>>(WihF, WhhF, bihF, bhhF,
                                                 WihB, WhhB, bihB, bhhB,
                                                 Wf, bias);
  }
  // phase 1: fused bidirectional LSTM + pooling
  lstm_dir_kernel<<<dim3(BATCH / TB, 2), 256, 0, stream>>>(x, emb, Wf, bias, pooled);

  // phase 2: classifier head
  lstm_fc_kernel<<<(BATCH * NCLS + 255) / 256, 256, 0, stream>>>(pooled, fcW, fcb, out);
}
